// SINGLEModel_32495722561690
// MI455X (gfx1250) — matrix operand order repaired and checked by reading, not compiled, not run on hardware
//
#include <hip/hip_runtime.h>
#include <hip/hip_bf16.h>
#include <math.h>

// ---------------------------------------------------------------------------
// Problem constants (match reference)
// ---------------------------------------------------------------------------
#define HEADS   8
#define DHEAD   16
#define EMBD    128
#define NBATCH  32
#define NNODE   500
#define NPOMO   500
#define NPAD    512           // N and P padded to 32 tiles of 16
#define KQ      160           // 129 (E+1) padded to 5 chunks of 32
#define BIGNEG  (-1e30f)
#define INV_SQRT_D   0.25f                    // 1/sqrt(16)
#define INV_SQRT_EMB 0.08838834764831845f     // 1/sqrt(128)
#define LOGIT_CLIP   10.0f

typedef __attribute__((ext_vector_type(16))) _Float16 v16h;
typedef __attribute__((ext_vector_type(8)))  _Float16 h8;
typedef __attribute__((ext_vector_type(8)))  float    v8f;

// ---------------------------------------------------------------------------
// WMMA: D = A(16x32 f16) * B(32x16 f16) + C(16x16 f32)
// ---------------------------------------------------------------------------
__device__ inline v8f wmma_f32_f16(v16h a, v16h b, v8f c) {
    return __builtin_amdgcn_wmma_f32_16x16x32_f16(
        false, a, false, b, (short)0, c, false, false);
}

// A 16x32 layout: lane holds row m = lane&15; one contiguous run per lane:
// e -> K = g*16+e, g = lane>>4.
// Source: row-major f16 tile, element (m,k) = base[m*stride + k]. Guard-free.
__device__ inline v16h pack_a_rows(const _Float16* base, int stride) {
    const int m = threadIdx.x & 15;
    const int g = (threadIdx.x >> 4) & 1;
    const _Float16* p = base + m * stride;
    h8 lo = *(const h8*)(p + g * 16);
    h8 hi = *(const h8*)(p + g * 16 + 8);
    v16h a;
#pragma unroll
    for (int e = 0; e < 8; ++e) { a[e] = lo[e]; a[8 + e] = hi[e]; }
    return a;
}

// A 16x32 with only K<16 real: g==0 half-wave loads, g==1 half-wave is zero.
__device__ inline v16h pack_a_k16(const _Float16* base, int stride) {
    const int m = threadIdx.x & 15;
    const int g = (threadIdx.x >> 4) & 1;
    v16h a = {};
    if (g == 0) {
        h8 lo = *(const h8*)(base + m * stride);
        h8 hi = *(const h8*)(base + m * stride + 8);
#pragma unroll
        for (int e = 0; e < 8; ++e) { a[e] = lo[e]; a[8 + e] = hi[e]; }
    }
    return a;
}

// B 32x16 layout: lane holds col n = lane&15; halves e -> K = g*16+e.
// Source is TRANSPOSED storage: element (k,n) = base[n*stride + k] -> one
// contiguous 16-half run per lane. Guard-free.
__device__ inline v16h pack_b_t(const _Float16* base, int stride) {
    const int n = threadIdx.x & 15;
    const int g = (threadIdx.x >> 4) & 1;
    const _Float16* p = base + n * stride + g * 16;
    h8 lo = *(const h8*)(p);
    h8 hi = *(const h8*)(p + 8);
    v16h b;
#pragma unroll
    for (int e = 0; e < 8; ++e) { b[e] = lo[e]; b[8 + e] = hi[e]; }
    return b;
}

// B 32x16 with only K<16 real: g==0 half-wave loads, g==1 half-wave is zero.
__device__ inline v16h pack_b_t_k16(const _Float16* base, int stride) {
    const int n = threadIdx.x & 15;
    const int g = (threadIdx.x >> 4) & 1;
    v16h b = {};
    if (g == 0) {
        const _Float16* p = base + n * stride;
        h8 lo = *(const h8*)(p);
        h8 hi = *(const h8*)(p + 8);
#pragma unroll
        for (int e = 0; e < 8; ++e) { b[e] = lo[e]; b[8 + e] = hi[e]; }
    }
    return b;
}

// C/D 16x16 f32 layout: element r -> row M = r + 8*(lane>>4), col N = lane&15.

__device__ inline float red_max16(float v) {
#pragma unroll
    for (int m = 1; m < 16; m <<= 1) v = fmaxf(v, __shfl_xor(v, m, 32));
    return v;
}
__device__ inline float red_sum16(float v) {
#pragma unroll
    for (int m = 1; m < 16; m <<= 1) v += __shfl_xor(v, m, 32);
    return v;
}

// ---------------------------------------------------------------------------
// Kernel 0: one-time f32 -> f16 conversions with zero padding (all guards here)
// ---------------------------------------------------------------------------
__global__ void __launch_bounds__(256)
cvt_kernel(const float* __restrict__ enc, const float* __restrict__ last,
           const float* __restrict__ attr, const float* __restrict__ Wq,
           const float* __restrict__ Wk, const float* __restrict__ Wv,
           const float* __restrict__ Wc, _Float16* __restrict__ encf16,
           _Float16* __restrict__ catf16, _Float16* __restrict__ Wqf16,
           _Float16* __restrict__ Wkf16, _Float16* __restrict__ Wvf16,
           _Float16* __restrict__ Wcf16) {
    const int tid = blockIdx.x * blockDim.x + threadIdx.x;
    const int nt = gridDim.x * blockDim.x;
    // weights 128x128
    for (int i = tid; i < EMBD * EMBD; i += nt) {
        Wkf16[i] = (_Float16)Wk[i];
        Wvf16[i] = (_Float16)Wv[i];
        Wcf16[i] = (_Float16)Wc[i];
    }
    // Wq 128x129 -> 128x160 zero padded
    for (int i = tid; i < EMBD * KQ; i += nt) {
        const int f = i / KQ, e = i - f * KQ;
        Wqf16[i] = (_Float16)((e <= EMBD) ? Wq[(size_t)f * (EMBD + 1) + e] : 0.f);
    }
    // enc [B,500,128] -> [B,512,128] zero padded
    for (int i = tid; i < NBATCH * NPAD * EMBD; i += nt) {
        const int b = i >> 16;                 // /(512*128)
        const int n = (i >> 7) & (NPAD - 1);
        const int e = i & 127;
        encf16[i] = (_Float16)((n < NNODE)
                        ? enc[((size_t)b * NNODE + n) * EMBD + e] : 0.f);
    }
    // cat(last,attr) [B,500,129] -> [B,512,160] zero padded
    for (int i = tid; i < NBATCH * NPAD * KQ; i += nt) {
        const int b = i / (NPAD * KQ);
        const int r = i - b * (NPAD * KQ);
        const int p = r / KQ, e = r - p * KQ;
        float v = 0.f;
        if (p < NPOMO) {
            if (e < EMBD)       v = last[((size_t)b * NPOMO + p) * EMBD + e];
            else if (e == EMBD) v = attr[(size_t)b * NPOMO + p];
        }
        catf16[i] = (_Float16)v;
    }
}

// ---------------------------------------------------------------------------
// Kernel 1: K/V projections.  K stored [B][H][512][16] (row-major n,d);
// V stored TRANSPOSED [B][H][16][512] so the w@V B-operand is contiguous.
// grid (NPAD/16, B), block 256; wave w owns features 16w..16w+15 (= head w).
// ---------------------------------------------------------------------------
__global__ void __launch_bounds__(256)
kv_proj_kernel(const _Float16* __restrict__ encf16,
               const _Float16* __restrict__ Wkf16,
               const _Float16* __restrict__ Wvf16,
               _Float16* __restrict__ kf16, _Float16* __restrict__ vtf16) {
    const int n0 = blockIdx.x * 16;
    const int b  = blockIdx.y;
    const int w  = threadIdx.x >> 5;
    const _Float16* arow = encf16 + ((size_t)b * NPAD + n0) * EMBD;
    v8f ck = {}, cv = {};
#pragma unroll
    for (int k0 = 0; k0 < EMBD; k0 += 32) {
        v16h a  = pack_a_rows(arow + k0, EMBD);
        v16h bk = pack_b_t(Wkf16 + (size_t)(w * 16) * EMBD + k0, EMBD);
        ck = wmma_f32_f16(a, bk, ck);
        v16h bv = pack_b_t(Wvf16 + (size_t)(w * 16) * EMBD + k0, EMBD);
        cv = wmma_f32_f16(a, bv, cv);
    }
    const int lane = threadIdx.x & 31;
    const int col = lane & 15, g = (lane >> 4) & 1;   // col = d, head = w
#pragma unroll
    for (int r = 0; r < 8; ++r) {
        const int row = r + 8 * g;                    // local node index
        kf16[(((size_t)(b * HEADS + w) * NPAD) + n0 + row) * DHEAD + col] =
            (_Float16)ck[r];
        vtf16[(((size_t)(b * HEADS + w) * DHEAD) + col) * NPAD + n0 + row] =
            (_Float16)cv[r];
    }
}

// ---------------------------------------------------------------------------
// Kernel 2: Q projection (K = 160).  grid (NPAD/16, B), block 256.
// ---------------------------------------------------------------------------
__global__ void __launch_bounds__(256)
q_proj_kernel(const _Float16* __restrict__ catf16,
              const _Float16* __restrict__ Wqf16, _Float16* __restrict__ qf16) {
    const int p0 = blockIdx.x * 16;
    const int b  = blockIdx.y;
    const int w  = threadIdx.x >> 5;
    const _Float16* arow = catf16 + ((size_t)b * NPAD + p0) * KQ;
    v8f c = {};
#pragma unroll
    for (int k0 = 0; k0 < KQ; k0 += 32) {
        v16h a  = pack_a_rows(arow + k0, KQ);
        v16h bb = pack_b_t(Wqf16 + (size_t)(w * 16) * KQ + k0, KQ);
        c = wmma_f32_f16(a, bb, c);
    }
    const int lane = threadIdx.x & 31;
    const int col = lane & 15, g = (lane >> 4) & 1;
#pragma unroll
    for (int r = 0; r < 8; ++r) {
        const int row = r + 8 * g;
        qf16[((size_t)b * NPAD + p0 + row) * EMBD + w * 16 + col] = (_Float16)c[r];
    }
}

// ---------------------------------------------------------------------------
// Kernel 3: fused MHA (flash-style online softmax) + combine with W_comb.
// grid (NPAD/16 p-tiles, B), block 256; wave h = head h.
// ---------------------------------------------------------------------------
__global__ void __launch_bounds__(256)
attn_kernel(const float* __restrict__ mask, const _Float16* __restrict__ Wcf16,
            const float* __restrict__ bc, const _Float16* __restrict__ qf16,
            const _Float16* __restrict__ kf16, const _Float16* __restrict__ vtf16,
            _Float16* __restrict__ mhf16) {
    __shared__ float    maskLds[16][NPAD];     // mask (+pad BIGNEG), 32 KB
    __shared__ _Float16 Pld[HEADS][16][32];    // per-wave softmax weights
    __shared__ _Float16 outAll[16][EMBD];      // gathered head outputs

    const int p0 = blockIdx.x * 16;
    const int b  = blockIdx.y;
    const int h  = threadIdx.x >> 5;
    const int lane = threadIdx.x & 31;
    const int col = lane & 15, g = (lane >> 4) & 1;

    // stage mask tile once (only guarded loads in this kernel)
    for (int i = threadIdx.x; i < 16 * NPAD; i += 256) {
        const int row = i >> 9, c = i & (NPAD - 1);
        const int p = p0 + row;
        float v = BIGNEG;
        if (c < NNODE)
            v = (p < NPOMO) ? mask[((size_t)b * NPOMO + p) * NNODE + c] : 0.f;
        maskLds[row][c] = v;
    }
    __syncthreads();

    v16h aq = pack_a_k16(qf16 + ((size_t)b * NPAD + p0) * EMBD + h * DHEAD, EMBD);

    const _Float16* kh  = kf16  + (size_t)(b * HEADS + h) * NPAD * DHEAD;
    const _Float16* vth = vtf16 + (size_t)(b * HEADS + h) * DHEAD * NPAD;

    v8f acc = {};
    float mrow[8], lrow[8];
#pragma unroll
    for (int r = 0; r < 8; ++r) { mrow[r] = BIGNEG; lrow[r] = 0.f; }

    for (int cchunk = 0; cchunk < NPAD / 32; ++cchunk) {
        const int n0 = cchunk * 32;
        v16h bk0 = pack_b_t_k16(kh + (size_t)n0 * DHEAD, DHEAD);
        v16h bk1 = pack_b_t_k16(kh + (size_t)(n0 + 16) * DHEAD, DHEAD);
        v8f z = {};
        v8f s0 = wmma_f32_f16(aq, bk0, z);
        v8f s1 = wmma_f32_f16(aq, bk1, z);

#pragma unroll
        for (int r = 0; r < 8; ++r) {
            const int row = r + 8 * g;
            float v0 = s0[r] * INV_SQRT_D + maskLds[row][n0 + col];
            float v1 = s1[r] * INV_SQRT_D + maskLds[row][n0 + 16 + col];

            float mt = red_max16(fmaxf(v0, v1));
            float mnew = fmaxf(mrow[r], mt);
            float alpha = __expf(mrow[r] - mnew);
            float e0 = __expf(v0 - mnew);
            float e1 = __expf(v1 - mnew);
            float rs = red_sum16(e0 + e1);
            lrow[r] = lrow[r] * alpha + rs;
            acc[r] *= alpha;
            mrow[r] = mnew;

            Pld[h][row][col]      = (_Float16)e0;
            Pld[h][row][16 + col] = (_Float16)e1;
        }
        __syncthreads();   // cross-lane repack of P into A layout
        v16h ap = pack_a_rows(&Pld[h][0][0], 32);
        v16h bv = pack_b_t(vth + n0, NPAD);    // B(K=node,N=d) = vt[d][n0+K]
        acc = wmma_f32_f16(ap, bv, acc);
        __syncthreads();   // protect Pld before next iteration's writes
    }

    // normalize, publish per-head output columns
#pragma unroll
    for (int r = 0; r < 8; ++r) {
        const int row = r + 8 * g;
        outAll[row][h * DHEAD + col] = (_Float16)(acc[r] / lrow[r]);
    }
    __syncthreads();

    // mh = outAll[16x128] @ W_comb^T + b_comb; wave w owns cols 16w..16w+15
    const int w = h;
    v8f c2 = {};
#pragma unroll
    for (int k0 = 0; k0 < EMBD; k0 += 32) {
        v16h a2 = pack_a_rows(&outAll[0][k0], EMBD);
        v16h b2 = pack_b_t(Wcf16 + (size_t)(w * 16) * EMBD + k0, EMBD);
        c2 = wmma_f32_f16(a2, b2, c2);
    }
    const float bias = bc[w * 16 + col];
#pragma unroll
    for (int r = 0; r < 8; ++r) {
        const int row = r + 8 * g;
        mhf16[((size_t)b * NPAD + p0 + row) * EMBD + w * 16 + col] =
            (_Float16)(c2[r] + bias);
    }
}

// ---------------------------------------------------------------------------
// Kernel 4: pointer scoring  probs = softmax(10*tanh(mh.enc^T/sqrt(E)) + mask)
// grid (NPAD/16 p-tiles, B), block 256; waves split the 32 n-tiles.
// ---------------------------------------------------------------------------
__global__ void __launch_bounds__(256)
pointer_kernel(const float* __restrict__ mask, const _Float16* __restrict__ mhf16,
               const _Float16* __restrict__ encf16, float* __restrict__ out) {
    __shared__ float sbuf[16][NPAD];           // clipped scores + mask, 32 KB
    __shared__ float mbuf[16][NPAD];           // staged mask, 32 KB
    __shared__ float rmax[16], rsum[16];

    const int p0 = blockIdx.x * 16;
    const int b  = blockIdx.y;
    const int w  = threadIdx.x >> 5;
    const int lane = threadIdx.x & 31;
    const int col = lane & 15, g = (lane >> 4) & 1;

    for (int i = threadIdx.x; i < 16 * NPAD; i += 256) {
        const int row = i >> 9, c = i & (NPAD - 1);
        const int p = p0 + row;
        float v = BIGNEG;
        if (c < NNODE)
            v = (p < NPOMO) ? mask[((size_t)b * NPOMO + p) * NNODE + c] : 0.f;
        mbuf[row][c] = v;
    }
    __syncthreads();

    // A operands (mh rows) packed once per wave, reused for all its n-tiles
    const _Float16* arow = mhf16 + ((size_t)b * NPAD + p0) * EMBD;
    v16h a[4];
#pragma unroll
    for (int kk = 0; kk < 4; ++kk) a[kk] = pack_a_rows(arow + kk * 32, EMBD);

    for (int t = w; t < NPAD / 16; t += 8) {
        const int n0 = t * 16;
        const _Float16* brow = encf16 + ((size_t)b * NPAD + n0) * EMBD;
        v8f s = {};
#pragma unroll
        for (int kk = 0; kk < 4; ++kk)
            s = wmma_f32_f16(a[kk], pack_b_t(brow + kk * 32, EMBD), s);
#pragma unroll
        for (int r = 0; r < 8; ++r) {
            const int row = r + 8 * g;
            const int n = n0 + col;
            sbuf[row][n] = LOGIT_CLIP * tanhf(s[r] * INV_SQRT_EMB) + mbuf[row][n];
        }
    }
    __syncthreads();

    {   // two-pass row softmax stats: 16 threads per row
        const int row = threadIdx.x >> 4;
        const int j = threadIdx.x & 15;
        float m = BIGNEG;
        for (int c = j; c < NPAD; c += 16) m = fmaxf(m, sbuf[row][c]);
        m = red_max16(m);
        float s = 0.f;
        for (int c = j; c < NPAD; c += 16) s += __expf(sbuf[row][c] - m);
        s = red_sum16(s);
        if (j == 0) { rmax[row] = m; rsum[row] = s; }
    }
    __syncthreads();

    for (int idx = threadIdx.x; idx < 16 * NNODE; idx += 256) {
        const int row = idx / NNODE, n = idx - row * NNODE;
        const int p = p0 + row;
        if (p < NPOMO)
            out[((size_t)b * NPOMO + p) * NNODE + n] =
                __expf(sbuf[row][n] - rmax[row]) / rsum[row];
    }
}

// ---------------------------------------------------------------------------
// Launch: 5 stream-ordered kernels; f16 intermediates in d_ws (~26 MB).
// ---------------------------------------------------------------------------
extern "C" void kernel_launch(void* const* d_in, const int* in_sizes, int n_in,
                              void* d_out, int out_size, void* d_ws, size_t ws_size,
                              hipStream_t stream) {
    const float* enc  = (const float*)d_in[0];  // [B,N,E]
    const float* last = (const float*)d_in[1];  // [B,P,E]
    const float* attr = (const float*)d_in[2];  // [B,P,1]
    const float* msk  = (const float*)d_in[3];  // [B,P,N]
    const float* Wq   = (const float*)d_in[4];  // [128,129]
    const float* Wk   = (const float*)d_in[5];  // [128,128]
    const float* Wv   = (const float*)d_in[6];  // [128,128]
    const float* Wc   = (const float*)d_in[7];  // [128,128]
    const float* bc   = (const float*)d_in[8];  // [128]
    float* out = (float*)d_out;

    const size_t MAT  = (size_t)NBATCH * NPAD * EMBD;          // 2,097,152
    const size_t CAT  = (size_t)NBATCH * NPAD * KQ;            // 2,621,440
    const size_t KV   = (size_t)NBATCH * HEADS * NPAD * DHEAD; // 2,097,152
    _Float16* encf16 = (_Float16*)d_ws;
    _Float16* catf16 = encf16 + MAT;
    _Float16* qf16   = catf16 + CAT;
    _Float16* kf16   = qf16 + MAT;
    _Float16* vtf16  = kf16 + KV;
    _Float16* mhf16  = vtf16 + KV;
    _Float16* Wqf16  = mhf16 + MAT;
    _Float16* Wkf16  = Wqf16 + (size_t)EMBD * KQ;
    _Float16* Wvf16  = Wkf16 + (size_t)EMBD * EMBD;
    _Float16* Wcf16  = Wvf16 + (size_t)EMBD * EMBD;
    // total ~= 13.2M halves ~= 26.4 MB

    dim3 grid(NPAD / 16, NBATCH);
    dim3 blk(256);
    cvt_kernel<<<dim3(512), blk, 0, stream>>>(enc, last, attr, Wq, Wk, Wv, Wc,
                                              encf16, catf16, Wqf16, Wkf16,
                                              Wvf16, Wcf16);
    kv_proj_kernel<<<grid, blk, 0, stream>>>(encf16, Wkf16, Wvf16, kf16, vtf16);
    q_proj_kernel<<<grid, blk, 0, stream>>>(catf16, Wqf16, qf16);
    attn_kernel<<<grid, blk, 0, stream>>>(msk, Wcf16, bc, qf16, kf16, vtf16, mhf16);
    pointer_kernel<<<grid, blk, 0, stream>>>(msk, mhf16, encf16, out);
}
